// swinu2net_61701500174570
// MI455X (gfx1250) — compile-verified, matrix operand order checked by reading, not run on hardware
//
#include <hip/hip_runtime.h>

typedef __attribute__((ext_vector_type(16))) _Float16 v16h;
typedef __attribute__((ext_vector_type(8)))  _Float16 v8h;
typedef __attribute__((ext_vector_type(8)))  float    v8f;
typedef __attribute__((ext_vector_type(4)))  unsigned int u32x4;
typedef __attribute__((ext_vector_type(8)))  int i32x8;
typedef __attribute__((ext_vector_type(4)))  int i32x4;

#define HWT   32768
#define CDIM  256
#define NROWS 65536   // B * HWT tokens (also windowed rows)

// ---------------------------------------------------------------------------
// helpers
// ---------------------------------------------------------------------------
static __device__ __forceinline__ v16h frag_ld(const _Float16* p) {
  // A/B 16-bit fragment: halves 0..7 = K[kbase..kbase+7], 8..15 = K[kbase+16..+23]
  v8h lo = *(const v8h*)(p);
  v8h hi = *(const v8h*)(p + 16);
  return __builtin_shufflevector(lo, hi, 0,1,2,3,4,5,6,7,8,9,10,11,12,13,14,15);
}

static __device__ __forceinline__ int region3(int g) {
  return g < 28 ? 0 : (g < 30 ? 1 : 2);
}

// ---------------------------------------------------------------------------
// TDM: DMA a [rows x 32half] tile (row stride = kstride elements) from global
// into LDS with +16B padding per 64B row => padded LDS row stride 40 halves.
// D# group0: count | lds_addr | global_addr | type=2
// D# group1: data_size=2B, pad_enable, pad_interval=16dw(code3), pad_amount=4dw(code3),
//            tensor_dim0=kstride, tensor_dim1=big, tile_dim0=32, tile_dim1=rows,
//            tensor_dim0_stride=kstride
// groups 2/3 zero (<=2D tensor). Must be issued from a single wave (scalar branch).
// This toolchain's builtin arity is 6: (u32x4, i32x8, i32x4, i32x4, i32x8, i32).
// ---------------------------------------------------------------------------
static __device__ __forceinline__ void tdm_load_tile32(const _Float16* gsrc,
                                                       unsigned lds_off,
                                                       int rows, int kstride) {
  unsigned long long ga = (unsigned long long)(size_t)gsrc;
  u32x4 g0;
  g0[0] = 1u;                                   // count=1 (valid user descriptor)
  g0[1] = lds_off;                              // LDS byte address
  g0[2] = (unsigned)ga;                         // global addr [31:0]
  g0[3] = (unsigned)(ga >> 32) | (2u << 30);    // global addr [56:32] | type=2
  unsigned t0 = (unsigned)kstride;              // tensor_dim0 (elements)
  unsigned t1 = 1u << 20;                       // tensor_dim1 (large; OOB only)
  unsigned long long s0 = (unsigned long long)kstride; // dim0 stride (elements)
  i32x8 g1;
  g1[0] = (int)((1u << 16) | (1u << 20) | (3u << 22) | (3u << 25));
  g1[1] = (int)((t0 & 0xffffu) << 16);          // abar addr=0 | tensor_dim0 lo
  g1[2] = (int)((t0 >> 16) | ((t1 & 0xffffu) << 16));
  g1[3] = (int)((t1 >> 16) | (32u << 16));      // tile_dim0 = 32 elements
  g1[4] = (int)(unsigned)rows;                  // tile_dim1 = rows; tile_dim2 = 0
  g1[5] = (int)(unsigned)(s0 & 0xffffffffu);
  g1[6] = (int)(unsigned)(s0 >> 32);            // dim1 stride unused (2D)
  g1[7] = 0;
  i32x4 z4 = (i32x4){};
  i32x8 z8 = (i32x8){};
  __builtin_amdgcn_tensor_load_to_lds(g0, g1, z4, z4, z8, 0);
}

// ---------------------------------------------------------------------------
// weight convert (+optional transpose): f32 W[K][N] -> f16 Wt[N][K]
// ---------------------------------------------------------------------------
__global__ __launch_bounds__(256)
void wprep_kernel(const float* __restrict__ w, _Float16* __restrict__ wt,
                  int K, int N, int transpose) {
  int i = blockIdx.x * 256 + threadIdx.x;
  if (i >= K * N) return;
  if (transpose) {
    int k = i / N, n = i - k * N;
    wt[(size_t)n * K + k] = (_Float16)w[i];
  } else {
    wt[i] = (_Float16)w[i];
  }
}

// ---------------------------------------------------------------------------
// input transpose: x[B][C][HWT] f32 -> resid[B*HWT][C] f32
// ---------------------------------------------------------------------------
__global__ __launch_bounds__(256)
void xin_kernel(const float* __restrict__ x, float* __restrict__ resid) {
  int idx = blockIdx.x * 256 + threadIdx.x;     // 16,777,216 total
  int b   = idx >> 23;
  int rem = idx & 8388607;
  int c   = rem >> 15;
  int pos = rem & 32767;
  resid[((size_t)b * HWT + pos) * CDIM + c] = x[idx];
}

// ---------------------------------------------------------------------------
// LayerNorm (+ optional shift-roll + window-partition gather), fp32 -> fp16
// one wave per output row; lane holds 8 channels
// ---------------------------------------------------------------------------
__global__ __launch_bounds__(256)
void ln_kernel(const float* __restrict__ x, const float* __restrict__ gam,
               const float* __restrict__ bet, _Float16* __restrict__ out,
               int shift, int windowed) {
  int row  = blockIdx.x * 8 + (threadIdx.x >> 5);
  int lane = threadIdx.x & 31;
  size_t src;
  if (windowed) {
    int wid = row >> 6, n = row & 63;
    int bb  = wid >> 9, widb = wid & 511;
    int wh = widb >> 6, wwc = (widb >> 3) & 7, wtc = widb & 7;
    int ih = n >> 4, iw = (n >> 2) & 3, it = n & 3;
    int gh = (wh * 4 + ih + shift) & 31;
    int gw = (wwc * 4 + iw + shift) & 31;
    int gt = (wtc * 4 + it + shift) & 31;
    src = (size_t)bb * HWT + ((gh * 32 + gw) * 32 + gt);
  } else {
    src = (size_t)row;
  }
  const float* xp = x + src * CDIM + lane * 8;
  float v[8];
  float s = 0.f;
#pragma unroll
  for (int i = 0; i < 8; ++i) { v[i] = xp[i]; s += v[i]; }
#pragma unroll
  for (int d = 1; d < 32; d <<= 1) s += __shfl_xor(s, d, 32);
  float mu = s * (1.0f / 256.0f);
  float q = 0.f;
#pragma unroll
  for (int i = 0; i < 8; ++i) { float e = v[i] - mu; q += e * e; }
#pragma unroll
  for (int d = 1; d < 32; d <<= 1) q += __shfl_xor(q, d, 32);
  float rs = rsqrtf(q * (1.0f / 256.0f) + 1e-5f);
  _Float16* op = out + (size_t)row * CDIM + lane * 8;
#pragma unroll
  for (int i = 0; i < 8; ++i)
    op[i] = (_Float16)((v[i] - mu) * rs * gam[lane * 8 + i] + bet[lane * 8 + i]);
}

// ---------------------------------------------------------------------------
// Generic WMMA GEMM: X[M,K] f16 @ Wt[N,K]^T + bias  (block tile 64x128,
// 8 waves, wave tile 32x32 = 2x2 x v_wmma_f32_16x16x32_f16, K-step 32).
// Tiles staged by TDM (tensor_load_to_lds) with double buffering; wave 0
// issues descriptors (scalar-branch gated: TDM ignores EXEC).
// modes: 0 store f16  1 gelu->f16  2 proj scatter-add resid  3 add resid
//        4 relu -> NCDHW f32 out
// ---------------------------------------------------------------------------
__global__ __launch_bounds__(256)
void gemm_wmma_kernel(const _Float16* __restrict__ X, const _Float16* __restrict__ Wt,
                      const float* __restrict__ bias, _Float16* __restrict__ outh,
                      float* __restrict__ resid, float* __restrict__ outf,
                      int M, int N, int K, int mode, int shift) {
  __shared__ _Float16 As[2][64 * 40];
  __shared__ _Float16 Bs[2][128 * 40];
  (void)M;
  const int tid  = threadIdx.x;
  const int lane = tid & 31;
  const int wave = tid >> 5;
  const int wm = wave >> 2, wn = wave & 3;
  const int bM = blockIdx.y * 64, bN = blockIdx.x * 128;
  const int hl = lane >> 4, l15 = lane & 15;
  const int kbase = hl * 8;
  const bool issuer = (__builtin_amdgcn_readfirstlane(wave) == 0);

  const unsigned aOff[2] = { (unsigned)(size_t)(&As[0][0]), (unsigned)(size_t)(&As[1][0]) };
  const unsigned bOff[2] = { (unsigned)(size_t)(&Bs[0][0]), (unsigned)(size_t)(&Bs[1][0]) };
  const _Float16* Abase = X  + (size_t)bM * K;
  const _Float16* Bbase = Wt + (size_t)bN * K;

  v8f acc[2][2];
#pragma unroll
  for (int i = 0; i < 2; ++i)
#pragma unroll
    for (int j = 0; j < 2; ++j) acc[i][j] = (v8f){};

  if (issuer) {                       // prologue: tile 0 -> buffer 0
    tdm_load_tile32(Abase, aOff[0], 64, K);
    tdm_load_tile32(Bbase, bOff[0], 128, K);
  }

  int cur = 0;
  for (int k0 = 0; k0 < K; k0 += 32) {
    if (issuer) {
      if (k0 + 32 < K) {              // prefetch next tile into other buffer
        tdm_load_tile32(Abase + k0 + 32, aOff[cur ^ 1], 64, K);
        tdm_load_tile32(Bbase + k0 + 32, bOff[cur ^ 1], 128, K);
        __builtin_amdgcn_s_wait_tensorcnt(2);   // current tile landed
      } else {
        __builtin_amdgcn_s_wait_tensorcnt(0);
      }
    }
    __syncthreads();                  // publish buffer `cur`

    const _Float16* Ac = &As[cur][0];
    const _Float16* Bc = &Bs[cur][0];
    v16h af[2], bf[2];
#pragma unroll
    for (int fm = 0; fm < 2; ++fm)
      af[fm] = frag_ld(Ac + (wm * 32 + fm * 16 + l15) * 40 + kbase);
#pragma unroll
    for (int fn = 0; fn < 2; ++fn)
      bf[fn] = frag_ld(Bc + (wn * 32 + fn * 16 + l15) * 40 + kbase);
#pragma unroll
    for (int fm = 0; fm < 2; ++fm)
#pragma unroll
      for (int fn = 0; fn < 2; ++fn)
        acc[fm][fn] = __builtin_amdgcn_wmma_f32_16x16x32_f16(
            false, af[fm], false, bf[fn], (short)0, acc[fm][fn], false, false);

    __syncthreads();                  // everyone done reading before overwrite
    cur ^= 1;
  }

#pragma unroll
  for (int fm = 0; fm < 2; ++fm)
#pragma unroll
    for (int fn = 0; fn < 2; ++fn) {
      int colN = bN + wn * 32 + fn * 16 + l15;
      float bb = bias[colN];
#pragma unroll
      for (int v = 0; v < 8; ++v) {
        int rowM  = bM + wm * 32 + fm * 16 + v + 8 * hl;
        float val = acc[fm][fn][v] + bb;
        if (mode == 0) {
          outh[(size_t)rowM * N + colN] = (_Float16)val;
        } else if (mode == 1) {
          val = 0.5f * val * (1.0f + erff(val * 0.70710678118f));
          outh[(size_t)rowM * N + colN] = (_Float16)val;
        } else if (mode == 2) {
          // window-reverse (+roll back) scatter-add into residual stream
          int wid = rowM >> 6, n = rowM & 63;
          int bb2 = wid >> 9, widb = wid & 511;
          int wh = widb >> 6, wwc = (widb >> 3) & 7, wtc = widb & 7;
          int ih = n >> 4, iw = (n >> 2) & 3, it = n & 3;
          int gh = (wh * 4 + ih + shift) & 31;
          int gw = (wwc * 4 + iw + shift) & 31;
          int gt = (wtc * 4 + it + shift) & 31;
          size_t tok = (size_t)bb2 * HWT + ((gh * 32 + gw) * 32 + gt);
          resid[tok * CDIM + colN] += val;
        } else if (mode == 3) {
          resid[(size_t)rowM * CDIM + colN] += val;
        } else { // 4: relu, scatter to [B][C][HWT] f32
          int b2 = rowM >> 15, pos = rowM & 32767;
          outf[((size_t)b2 * CDIM + colN) * HWT + pos] = fmaxf(val, 0.0f);
        }
      }
    }
}

// ---------------------------------------------------------------------------
// Windowed attention: one block (4 waves) per (window, head).
// Q/K head-slices DMA'd by TDM (row stride 768, padded LDS); V transposed
// manually. S = softmax(QK^T*scale + rpb + mask); O = S V; all via WMMA.
// ---------------------------------------------------------------------------
__global__ __launch_bounds__(128)
void attn_kernel(const _Float16* __restrict__ qkv, const float* __restrict__ rpb,
                 _Float16* __restrict__ o, int shift) {
  __shared__ _Float16 Qs[64 * 40];
  __shared__ _Float16 Ks[64 * 40];
  __shared__ _Float16 Vt[32 * 72];   // transposed: Vt[d][j]
  __shared__ _Float16 Ps[64 * 72];
  const int blk  = blockIdx.x;
  const int head = blk & 7;
  const int wid  = blk >> 3;
  const int tid  = threadIdx.x;
  const int lane = tid & 31;
  const int wave = tid >> 5;
  const int hl = lane >> 4, l15 = lane & 15;
  const int kbase = hl * 8;

  const _Float16* wbase = qkv + (size_t)wid * 64 * 768 + head * 32;
  if (__builtin_amdgcn_readfirstlane(wave) == 0) {
    tdm_load_tile32(wbase,       (unsigned)(size_t)(&Qs[0]), 64, 768);  // Q
    tdm_load_tile32(wbase + 256, (unsigned)(size_t)(&Ks[0]), 64, 768);  // K
    __builtin_amdgcn_s_wait_tensorcnt(0);
  }
  { // stage V transposed (all threads, concurrent with TDM)
    int row = tid >> 1, off = (tid & 1) * 16;
    const _Float16* vb = wbase + (size_t)row * 768 + 512 + off;
    v8h v0 = *(const v8h*)(vb);
    v8h v1 = *(const v8h*)(vb + 8);
#pragma unroll
    for (int i = 0; i < 8; ++i) {
      Vt[(off + i) * 72 + row]     = v0[i];
      Vt[(off + 8 + i) * 72 + row] = v1[i];
    }
  }
  __syncthreads();

  const int i0 = wave * 16;
  v8f sc[4];
  {
    v16h aq = frag_ld(Qs + (i0 + l15) * 40 + kbase);
#pragma unroll
    for (int t = 0; t < 4; ++t) {
      v16h bk = frag_ld(Ks + (t * 16 + l15) * 40 + kbase);
      v8f z = (v8f){};
      sc[t] = __builtin_amdgcn_wmma_f32_16x16x32_f16(false, aq, false, bk,
                                                     (short)0, z, false, false);
    }
  }

  // scale + relative position bias + shifted-window mask (analytic)
  const int widb = wid & 511;
  const int wh = widb >> 6, wwc = (widb >> 3) & 7, wtc = widb & 7;
#pragma unroll
  for (int t = 0; t < 4; ++t) {
#pragma unroll
    for (int v = 0; v < 8; ++v) {
      int i = i0 + v + 8 * hl;
      int j = t * 16 + l15;
      int ih = i >> 4, iw = (i >> 2) & 3, it = i & 3;
      int jh = j >> 4, jw = (j >> 2) & 3, jt = j & 3;
      int idx = (ih - jh + 3) * 49 + (iw - jw + 3) * 7 + (it - jt + 3);
      float s = sc[t][v] * 0.17677669f + rpb[(size_t)idx * 8 + head];
      if (shift) {
        int ri = region3(wh * 4 + ih) * 9 + region3(wwc * 4 + iw) * 3 + region3(wtc * 4 + it);
        int rj = region3(wh * 4 + jh) * 9 + region3(wwc * 4 + jw) * 3 + region3(wtc * 4 + jt);
        if (ri != rj) s -= 100.0f;
      }
      sc[t][v] = s;
    }
  }

  // row softmax: reduce across 4 col-tiles + 16 lanes of each half-wave
#pragma unroll
  for (int v = 0; v < 8; ++v) {
    float m = fmaxf(fmaxf(sc[0][v], sc[1][v]), fmaxf(sc[2][v], sc[3][v]));
#pragma unroll
    for (int d = 1; d < 16; d <<= 1) m = fmaxf(m, __shfl_xor(m, d, 32));
    float s = 0.f;
#pragma unroll
    for (int t = 0; t < 4; ++t) { float e = __expf(sc[t][v] - m); sc[t][v] = e; s += e; }
#pragma unroll
    for (int d = 1; d < 16; d <<= 1) s += __shfl_xor(s, d, 32);
    float inv = 1.0f / s;
#pragma unroll
    for (int t = 0; t < 4; ++t) sc[t][v] *= inv;
  }

  // P -> LDS (reshape C-layout into A-fragment layout)
#pragma unroll
  for (int t = 0; t < 4; ++t)
#pragma unroll
    for (int v = 0; v < 8; ++v)
      Ps[(i0 + v + 8 * hl) * 72 + t * 16 + l15] = (_Float16)sc[t][v];
  __syncthreads();

  // O = P @ V  (K=64: two WMMA k-steps per 16-col output tile)
#pragma unroll
  for (int fn = 0; fn < 2; ++fn) {
    v8f accO = (v8f){};
#pragma unroll
    for (int ks = 0; ks < 2; ++ks) {
      v16h ap = frag_ld(Ps + (i0 + l15) * 72 + ks * 32 + kbase);
      v16h bv = frag_ld(Vt + (fn * 16 + l15) * 72 + ks * 32 + kbase);
      accO = __builtin_amdgcn_wmma_f32_16x16x32_f16(false, ap, false, bv,
                                                    (short)0, accO, false, false);
    }
#pragma unroll
    for (int v = 0; v < 8; ++v) {
      int i = i0 + v + 8 * hl;
      o[(size_t)(wid * 64 + i) * CDIM + head * 32 + fn * 16 + l15] = (_Float16)accO[v];
    }
  }
}

// ---------------------------------------------------------------------------
// depthwise 3x3x3 'SAME' conv, channel-last residual -> f16 activations
// ---------------------------------------------------------------------------
__global__ __launch_bounds__(256)
void dwconv_kernel(const float* __restrict__ resid, const float* __restrict__ w,
                   _Float16* __restrict__ out) {
  int pos = blockIdx.x;          // 0..65535 = b*32768 + p
  int c   = threadIdx.x;
  int b = pos >> 15, p = pos & 32767;
  int h = p >> 10, wv = (p >> 5) & 31, t = p & 31;
  float s = 0.f;
#pragma unroll
  for (int dh = -1; dh <= 1; ++dh)
#pragma unroll
    for (int dw2 = -1; dw2 <= 1; ++dw2)
#pragma unroll
      for (int dt = -1; dt <= 1; ++dt) {
        int hh = h + dh, ww = wv + dw2, tt = t + dt;
        if (hh >= 0 && hh < 32 && ww >= 0 && ww < 32 && tt >= 0 && tt < 32) {
          float wt = w[c * 27 + (dh + 1) * 9 + (dw2 + 1) * 3 + (dt + 1)];
          s += wt * resid[((size_t)b * HWT + ((hh * 32 + ww) * 32 + tt)) * CDIM + c];
        }
      }
  out[(size_t)pos * CDIM + c] = (_Float16)s;
}

// ---------------------------------------------------------------------------
// host orchestration
// ---------------------------------------------------------------------------
extern "C" void kernel_launch(void* const* d_in, const int* in_sizes, int n_in,
                              void* d_out, int out_size, void* d_ws, size_t ws_size,
                              hipStream_t stream) {
  (void)in_sizes; (void)n_in; (void)out_size; (void)ws_size;
  const float* x     = (const float*)d_in[0];
  const float* g1    = (const float*)d_in[1];
  const float* b1    = (const float*)d_in[2];
  const float* qkv_w = (const float*)d_in[3];
  const float* qkv_b = (const float*)d_in[4];
  const float* proj_w= (const float*)d_in[5];
  const float* proj_b= (const float*)d_in[6];
  const float* rpb   = (const float*)d_in[7];
  const float* g2    = (const float*)d_in[8];
  const float* b2    = (const float*)d_in[9];
  const float* fc1_w = (const float*)d_in[10];
  const float* fc1_b = (const float*)d_in[11];
  const float* fc2_w = (const float*)d_in[12];
  const float* fc2_b = (const float*)d_in[13];
  const float* dw_w  = (const float*)d_in[14];
  const float* pw_w  = (const float*)d_in[15];
  const float* pw_b  = (const float*)d_in[16];

  char* ws = (char*)d_ws;
  float*    resid = (float*)(ws);                                  // 64 MB
  _Float16* act   = (_Float16*)(ws + (size_t)64  * 1024 * 1024);   // 32 MB
  _Float16* big   = (_Float16*)(ws + (size_t)96  * 1024 * 1024);   // 128 MB
  _Float16* wt    = (_Float16*)(ws + (size_t)224 * 1024 * 1024);   // ~6.2 MB
  const size_t PB = 786432;   // f16 weights per block

  // one-shot weight conversion (re-done every call: deterministic)
  for (int blk = 0; blk < 4; ++blk) {
    wprep_kernel<<<768,  256, 0, stream>>>(qkv_w + (size_t)blk * 196608, wt + blk * PB,            256, 768,  1);
    wprep_kernel<<<256,  256, 0, stream>>>(proj_w + (size_t)blk * 65536, wt + blk * PB + 196608,   256, 256,  1);
    wprep_kernel<<<1024, 256, 0, stream>>>(fc1_w  + (size_t)blk * 262144, wt + blk * PB + 262144,  256, 1024, 1);
    wprep_kernel<<<1024, 256, 0, stream>>>(fc2_w  + (size_t)blk * 262144, wt + blk * PB + 524288, 1024, 256,  1);
  }
  wprep_kernel<<<256, 256, 0, stream>>>(pw_w, wt + 4 * PB, 256, 256, 0); // pw_w[cout][cin] is already [N][K]

  xin_kernel<<<65536, 256, 0, stream>>>(x, resid);

  for (int blk = 0; blk < 4; ++blk) {
    int shift = (blk & 1) ? 2 : 0;
    const _Float16* qkvT = wt + blk * PB;
    const _Float16* projT = qkvT + 196608;
    const _Float16* fc1T  = projT + 65536;
    const _Float16* fc2T  = fc1T + 262144;

    // LN1 + roll + window partition -> act (windowed rows)
    ln_kernel<<<NROWS / 8, 256, 0, stream>>>(resid, g1 + blk * 256, b1 + blk * 256, act, shift, 1);
    // QKV gemm -> big [65536, 768]
    gemm_wmma_kernel<<<dim3(6, 1024), 256, 0, stream>>>(act, qkvT, qkv_b + blk * 768,
        big, nullptr, nullptr, NROWS, 768, 256, 0, 0);
    // windowed attention -> act [65536, 256]
    attn_kernel<<<8192, 128, 0, stream>>>(big, rpb + (size_t)blk * 343 * 8, act, shift);
    // proj gemm, window-reverse scatter-add into residual
    gemm_wmma_kernel<<<dim3(2, 1024), 256, 0, stream>>>(act, projT, proj_b + blk * 256,
        nullptr, resid, nullptr, NROWS, 256, 256, 2, shift);
    // LN2 -> act (token order)
    ln_kernel<<<NROWS / 8, 256, 0, stream>>>(resid, g2 + blk * 256, b2 + blk * 256, act, 0, 0);
    // fc1 + exact GELU -> big [65536, 1024]
    gemm_wmma_kernel<<<dim3(8, 1024), 256, 0, stream>>>(act, fc1T, fc1_b + blk * 1024,
        big, nullptr, nullptr, NROWS, 1024, 256, 1, 0);
    // fc2, residual add
    gemm_wmma_kernel<<<dim3(2, 1024), 256, 0, stream>>>(big, fc2T, fc2_b + blk * 256,
        nullptr, resid, nullptr, NROWS, 256, 1024, 3, 0);
  }

  // depthwise 3x3x3 -> act f16
  dwconv_kernel<<<65536, 256, 0, stream>>>(resid, dw_w, act);
  // pointwise conv as gemm, +bias, relu, scatter to NCDHW f32 output
  gemm_wmma_kernel<<<dim3(2, 1024), 256, 0, stream>>>(act, wt + 4 * PB, pw_b,
      nullptr, nullptr, (float*)d_out, NROWS, 256, 256, 4, 0);
}
